// LSTMPredictor_25563645346742
// MI455X (gfx1250) — matrix operand order repaired and checked by reading, not compiled, not run on hardware
//
#include <hip/hip_runtime.h>

// Problem constants (match reference setup)
#define B_  128
#define T_  256
#define D_  64
#define H_  1024
#define F_  64
#define TF_ (T_ + F_)          // 320
#define NWG 64
#define NTHREADS 256
#define KT0 34                 // (D_+H_)/32  K-tiles for layer-0 fused input
#define KT1 64                 // (2*H_)/32   K-tiles for layer-1 fused input
#define KTO 32                 // H_/32       K-tiles for output projection

typedef __bf16 bf16;
typedef __attribute__((ext_vector_type(16))) __bf16 v16bf;
typedef __attribute__((ext_vector_type(8)))  float  v8f;

// ---- WMMA fragment addressing (contiguous k order, k = 16*(lane/16) + slot; 16-bit 16x32 A / 32x16 B) ----
// One fragment block = 32 lanes x 16 halfs (1024 B). Stored lane-major so each
// lane loads its 16 bf16 values with one contiguous 32-byte load.
__device__ __forceinline__ unsigned a_off(int m, int k) {   // A: (m in 0..15, k in 0..31)
  unsigned lane = (unsigned)((m & 15) + (((k >> 4) & 1) << 4));
  unsigned slot = (unsigned)(k & 15);
  return (lane << 4) | slot;
}
__device__ __forceinline__ unsigned b_off(int k, int n) {   // B: (k in 0..31, n in 0..15)
  unsigned lane = (unsigned)((n & 15) + (((k >> 4) & 1) << 4));
  unsigned slot = (unsigned)(k & 15);
  return (lane << 4) | slot;
}

__device__ __forceinline__ v8f wmma_bf16(v16bf a, v16bf b, v8f c) {
  return __builtin_amdgcn_wmma_f32_16x16x32_bf16(false, a, false, b, (short)0, c, false, false);
}

__device__ __forceinline__ float sigmoid_(float x) { return 1.0f / (1.0f + __expf(-x)); }
__device__ __forceinline__ float tanh_(float x) {
  float e = __expf(2.0f * x);
  return (e - 1.0f) / (e + 1.0f);
}

// Sense-reversal grid barrier (persistent kernel, agent scope).
__device__ __forceinline__ void grid_sync(unsigned* bar) {
  __threadfence();
  __syncthreads();
  if (threadIdx.x == 0) {
    unsigned g = __hip_atomic_load(bar + 1, __ATOMIC_RELAXED, __HIP_MEMORY_SCOPE_AGENT);
    unsigned old = __hip_atomic_fetch_add(bar, 1u, __ATOMIC_ACQ_REL, __HIP_MEMORY_SCOPE_AGENT);
    if (old == (unsigned)(NWG - 1)) {
      __hip_atomic_store(bar, 0u, __ATOMIC_RELAXED, __HIP_MEMORY_SCOPE_AGENT);
      __hip_atomic_fetch_add(bar + 1, 1u, __ATOMIC_RELEASE, __HIP_MEMORY_SCOPE_AGENT);
    } else {
      while (__hip_atomic_load(bar + 1, __ATOMIC_ACQUIRE, __HIP_MEMORY_SCOPE_AGENT) == g)
        __builtin_amdgcn_s_sleep(2);
    }
  }
  __syncthreads();
  __threadfence();
}

// ---------------------------------------------------------------------------
// Prologue: fuse + convert weights to bf16 in WMMA B-fragment layout, fuse biases.
// W0 fused: [4096 x 1088] = [Wih0 | Whh0];  W1 fused: [4096 x 2048] = [Wih1 | Whh1]
// ---------------------------------------------------------------------------
__global__ void lstm_prep_kernel(const float* Wih0, const float* Whh0,
                                 const float* bih0, const float* bhh0,
                                 const float* Wih1, const float* Whh1,
                                 const float* bih1, const float* bhh1,
                                 const float* Wout,
                                 bf16* w0f, bf16* w1f, bf16* wof,
                                 float* b0f, float* b1f) {
  const long tid = (long)blockIdx.x * blockDim.x + threadIdx.x;
  const long stride = (long)gridDim.x * blockDim.x;
  const int K0 = D_ + H_;          // 1088
  const int K1 = 2 * H_;           // 2048
  // layer-0 fused weights
  for (long i = tid; i < (long)(4 * H_) * K0; i += stride) {
    int n = (int)(i / K0), k = (int)(i % K0);
    float v = (k < D_) ? Wih0[(long)n * D_ + k] : Whh0[(long)n * H_ + (k - D_)];
    w0f[((size_t)((n >> 4) * KT0 + (k >> 5)) << 9) + b_off(k & 31, n)] = (bf16)v;
  }
  // layer-1 fused weights
  for (long i = tid; i < (long)(4 * H_) * K1; i += stride) {
    int n = (int)(i >> 11), k = (int)(i & (K1 - 1));
    float v = (k < H_) ? Wih1[(long)n * H_ + k] : Whh1[(long)n * H_ + (k - H_)];
    w1f[((size_t)((n >> 4) * KT1 + (k >> 5)) << 9) + b_off(k & 31, n)] = (bf16)v;
  }
  // output projection  y = top @ Wout^T  ->  B[k][n] = Wout[n][k]
  for (long i = tid; i < (long)D_ * H_; i += stride) {
    int n = (int)(i >> 10), k = (int)(i & (H_ - 1));
    w0f += 0; // no-op
    wof[((size_t)((n >> 4) * KTO + (k >> 5)) << 9) + b_off(k & 31, n)] = (bf16)Wout[(long)n * H_ + k];
  }
  // fused biases
  for (long i = tid; i < 4 * H_; i += stride) {
    b0f[i] = bih0[i] + bhh0[i];
    b1f[i] = bih1[i] + bhh1[i];
  }
}

// ---------------------------------------------------------------------------
// Init: stage x_0 + h0 into fragment buffers, copy c state, reset barrier.
// actA buffer (layer-0 input)  : K-tiles 0..1  = x/y,  2..33  = h(layer0)
// actB buffer (layer-1 input)  : K-tiles 0..31 = h0new, 32..63 = h(layer1)
// ---------------------------------------------------------------------------
__global__ void lstm_init_kernel(const float* x, const float* h0in, const float* c0in,
                                 bf16* actA, bf16* actB, float* c0, float* c1,
                                 unsigned* bar) {
  const long tid = (long)blockIdx.x * blockDim.x + threadIdx.x;
  const long stride = (long)gridDim.x * blockDim.x;
  if (tid == 0) { bar[0] = 0u; bar[1] = 0u; }
  const int K0 = D_ + H_;
  // actA[parity 0]: (m,k) bijection onto the fragment buffer
  for (long i = tid; i < (long)B_ * K0; i += stride) {
    int m = (int)(i / K0), k = (int)(i % K0);
    float v = (k < D_) ? x[(long)m * T_ * D_ + k] : h0in[(long)m * H_ + (k - D_)];
    actA[((size_t)((k >> 5) * 8 + (m >> 4)) << 9) + a_off(m & 15, k & 31)] = (bf16)v;
  }
  // actB[parity 0]: zero K-tiles 0..31, h(layer1) into K-tiles 32..63
  for (long i = tid; i < (long)32 * 8 * 512; i += stride) actB[i] = (bf16)0.0f;
  for (long i = tid; i < (long)B_ * H_; i += stride) {
    int m = (int)(i >> 10), k = (int)(i & (H_ - 1));
    actB[((size_t)((32 + (k >> 5)) * 8 + (m >> 4)) << 9) + a_off(m & 15, k & 31)] =
        (bf16)h0in[(size_t)B_ * H_ + i];
  }
  for (long i = tid; i < (long)B_ * H_; i += stride) {
    c0[i] = c0in[i];
    c1[i] = c0in[(size_t)B_ * H_ + i];
  }
}

// ---------------------------------------------------------------------------
// Persistent LSTM kernel: 64 WGs x 256 threads (8 waves). WG `wg` owns hidden
// columns [wg*16, wg*16+16) across all 4 gate blocks; wave w owns M-tile w.
// ---------------------------------------------------------------------------
__launch_bounds__(NTHREADS)
__global__ void lstm_persistent_kernel(const float* x, const float* bout, float* out,
                                       const bf16* __restrict__ w0f,
                                       const bf16* __restrict__ w1f,
                                       const bf16* __restrict__ wof,
                                       const float* __restrict__ b0f,
                                       const float* __restrict__ b1f,
                                       bf16* actA, bf16* actB, float* c0, float* c1,
                                       unsigned* bar) {
  const int wg   = blockIdx.x;
  const int wave = threadIdx.x >> 5;      // 0..7 == M-tile
  const int lane = threadIdx.x & 31;
  const int col  = wg * 16 + (lane & 15); // hidden unit owned by this lane

  // hoist fused biases (constant across steps)
  const float bi0 = b0f[col], bf0 = b0f[H_ + col], bg0 = b0f[2 * H_ + col], bo0 = b0f[3 * H_ + col];
  const float bi1 = b1f[col], bf1 = b1f[H_ + col], bg1 = b1f[2 * H_ + col], bo1 = b1f[3 * H_ + col];
  const float by  = bout[((wg & 3) << 4) + (lane & 15)];   // used only by wg<32, wave 0

  const size_t ACTA_SZ = (size_t)KT0 * 8 * 512;   // halfs per parity buffer
  const size_t ACTB_SZ = (size_t)KT1 * 8 * 512;

  for (int step = 0; step < TF_; ++step) {
    const int p = step & 1, q = p ^ 1;
    bf16* actAp = actA + (size_t)p * ACTA_SZ;
    bf16* actAq = actA + (size_t)q * ACTA_SZ;
    bf16* actBp = actB + (size_t)p * ACTB_SZ;
    bf16* actBq = actB + (size_t)q * ACTB_SZ;

    // ---------------- phase 1: layer 0 gates + cell ----------------
    {
      v8f acc0 = {}, acc1 = {}, acc2 = {}, acc3 = {};
      const bf16* aP  = actAp + ((size_t)wave << 9) + ((size_t)lane << 4);
      const bf16* bP0 = w0f + (((size_t)(0 * 64 + wg) * KT0) << 9) + ((size_t)lane << 4);
      const bf16* bP1 = w0f + (((size_t)(1 * 64 + wg) * KT0) << 9) + ((size_t)lane << 4);
      const bf16* bP2 = w0f + (((size_t)(2 * 64 + wg) * KT0) << 9) + ((size_t)lane << 4);
      const bf16* bP3 = w0f + (((size_t)(3 * 64 + wg) * KT0) << 9) + ((size_t)lane << 4);
      #pragma unroll 2
      for (int kt = 0; kt < KT0; ++kt) {
        v16bf a = *(const v16bf*)aP; aP += 8 * 512;
        acc0 = wmma_bf16(a, *(const v16bf*)bP0, acc0); bP0 += 512;
        acc1 = wmma_bf16(a, *(const v16bf*)bP1, acc1); bP1 += 512;
        acc2 = wmma_bf16(a, *(const v16bf*)bP2, acc2); bP2 += 512;
        acc3 = wmma_bf16(a, *(const v16bf*)bP3, acc3); bP3 += 512;
      }
      #pragma unroll
      for (int r = 0; r < 8; ++r) {
        int m = 16 * wave + r + 8 * (lane >> 4);
        size_t cidx = (size_t)m * H_ + col;
        float gi = sigmoid_(acc0[r] + bi0);
        float gf = sigmoid_(acc1[r] + bf0);
        float gg = tanh_(acc2[r] + bg0);
        float go = sigmoid_(acc3[r] + bo0);
        float cn = gf * c0[cidx] + gi * gg;
        float hn = go * tanh_(cn);
        c0[cidx] = cn;
        bf16 hb = (bf16)hn;
        // h0_new as A-fragments: layer-1 input (this step) + layer-0 input (next step)
        actBp[((size_t)((col >> 5) * 8 + wave) << 9)        + a_off(m & 15, col & 31)] = hb;
        actAq[((size_t)((2 + (col >> 5)) * 8 + wave) << 9)  + a_off(m & 15, col & 31)] = hb;
      }
    }
    grid_sync(bar);

    // ---------------- phase 2: layer 1 gates + cell ----------------
    {
      v8f acc0 = {}, acc1 = {}, acc2 = {}, acc3 = {};
      const bf16* aP  = actBp + ((size_t)wave << 9) + ((size_t)lane << 4);
      const bf16* bP0 = w1f + (((size_t)(0 * 64 + wg) * KT1) << 9) + ((size_t)lane << 4);
      const bf16* bP1 = w1f + (((size_t)(1 * 64 + wg) * KT1) << 9) + ((size_t)lane << 4);
      const bf16* bP2 = w1f + (((size_t)(2 * 64 + wg) * KT1) << 9) + ((size_t)lane << 4);
      const bf16* bP3 = w1f + (((size_t)(3 * 64 + wg) * KT1) << 9) + ((size_t)lane << 4);
      #pragma unroll 2
      for (int kt = 0; kt < KT1; ++kt) {
        v16bf a = *(const v16bf*)aP; aP += 8 * 512;
        acc0 = wmma_bf16(a, *(const v16bf*)bP0, acc0); bP0 += 512;
        acc1 = wmma_bf16(a, *(const v16bf*)bP1, acc1); bP1 += 512;
        acc2 = wmma_bf16(a, *(const v16bf*)bP2, acc2); bP2 += 512;
        acc3 = wmma_bf16(a, *(const v16bf*)bP3, acc3); bP3 += 512;
      }
      #pragma unroll
      for (int r = 0; r < 8; ++r) {
        int m = 16 * wave + r + 8 * (lane >> 4);
        size_t cidx = (size_t)m * H_ + col;
        float gi = sigmoid_(acc0[r] + bi1);
        float gf = sigmoid_(acc1[r] + bf1);
        float gg = tanh_(acc2[r] + bg1);
        float go = sigmoid_(acc3[r] + bo1);
        float cn = gf * c1[cidx] + gi * gg;
        float hn = go * tanh_(cn);
        c1[cidx] = cn;
        // h1_new -> actB[q] K-tiles 32..63 (layer-1 recurrent input, next step; y input this step)
        actBq[((size_t)((32 + (col >> 5)) * 8 + wave) << 9) + a_off(m & 15, col & 31)] = (bf16)hn;
      }
    }
    grid_sync(bar);

    // ---------------- phase 3: y = top @ Wout^T + bout ----------------
    if (wg < 32 && wave == 0) {
      const int mt = wg >> 2, nt = wg & 3;
      v8f acc = {};
      const bf16* aP = actBq + (((size_t)(32 * 8 + mt)) << 9) + ((size_t)lane << 4);
      const bf16* bP = wof + (((size_t)nt * KTO) << 9) + ((size_t)lane << 4);
      #pragma unroll 2
      for (int kt = 0; kt < KTO; ++kt) {
        v16bf a = *(const v16bf*)aP; aP += 8 * 512;
        acc = wmma_bf16(a, *(const v16bf*)bP, acc); bP += 512;
      }
      #pragma unroll
      for (int r = 0; r < 8; ++r) {
        int m = mt * 16 + r + 8 * (lane >> 4);
        int n = nt * 16 + (lane & 15);
        float y = acc[r] + by;
        out[(size_t)m * (TF_ * D_) + (size_t)step * D_ + n] = y;
        if (step + 1 >= T_ && step + 1 < TF_) {   // autoregressive feedback
          actAq[((size_t)((n >> 5) * 8 + (m >> 4)) << 9) + a_off(m & 15, n & 31)] = (bf16)y;
        }
      }
    }
    // stage x[:, step+1, :] for the next teacher-forced step (all WGs help)
    if (step + 1 < T_) {
      for (int i = (int)threadIdx.x + wg * NTHREADS; i < B_ * D_; i += NWG * NTHREADS) {
        int m = i >> 6, d = i & 63;
        float v = x[(size_t)m * T_ * D_ + (size_t)(step + 1) * D_ + d];
        actAq[((size_t)((d >> 5) * 8 + (m >> 4)) << 9) + a_off(m & 15, d & 31)] = (bf16)v;
      }
    }
    grid_sync(bar);
  }
}

extern "C" void kernel_launch(void* const* d_in, const int* in_sizes, int n_in,
                              void* d_out, int out_size, void* d_ws, size_t ws_size,
                              hipStream_t stream) {
  (void)in_sizes; (void)n_in; (void)out_size; (void)ws_size;
  const float* x    = (const float*)d_in[0];
  const float* h0in = (const float*)d_in[1];
  const float* c0in = (const float*)d_in[2];
  /* d_in[3] = future_steps (fixed 64) */
  const float* Wih0 = (const float*)d_in[4];
  const float* Whh0 = (const float*)d_in[5];
  const float* bih0 = (const float*)d_in[6];
  const float* bhh0 = (const float*)d_in[7];
  const float* Wih1 = (const float*)d_in[8];
  const float* Whh1 = (const float*)d_in[9];
  const float* bih1 = (const float*)d_in[10];
  const float* bhh1 = (const float*)d_in[11];
  const float* Wout = (const float*)d_in[12];
  const float* bout = (const float*)d_in[13];
  float* out = (float*)d_out;

  // Workspace carve-up (~28.5 MB)
  char* w = (char*)d_ws;
  bf16* w0f = (bf16*)w;      w += (size_t)256 * KT0 * 1024;       // 8.9 MB
  bf16* w1f = (bf16*)w;      w += (size_t)256 * KT1 * 1024;       // 16.8 MB
  bf16* wof = (bf16*)w;      w += (size_t)4 * KTO * 1024;         // 128 KB
  float* b0f = (float*)w;    w += (size_t)4 * H_ * 4;
  float* b1f = (float*)w;    w += (size_t)4 * H_ * 4;
  bf16* actA = (bf16*)w;     w += (size_t)2 * KT0 * 8 * 1024;     // 544 KB
  bf16* actB = (bf16*)w;     w += (size_t)2 * KT1 * 8 * 1024;     // 1 MB
  float* c0 = (float*)w;     w += (size_t)B_ * H_ * 4;
  float* c1 = (float*)w;     w += (size_t)B_ * H_ * 4;
  unsigned* bar = (unsigned*)w;

  lstm_prep_kernel<<<2048, NTHREADS, 0, stream>>>(Wih0, Whh0, bih0, bhh0,
                                                  Wih1, Whh1, bih1, bhh1, Wout,
                                                  w0f, w1f, wof, b0f, b1f);
  lstm_init_kernel<<<512, NTHREADS, 0, stream>>>(x, h0in, c0in, actA, actB, c0, c1, bar);
  lstm_persistent_kernel<<<NWG, NTHREADS, 0, stream>>>(x, bout, out,
                                                       w0f, w1f, wof, b0f, b1f,
                                                       actA, actB, c0, c1, bar);
}
